// BitNetModel_16054587753066
// MI455X (gfx1250) — 8-bit matrix operand order repaired and checked by reading, compiled, not run on hardware
//
#include <hip/hip_runtime.h>
#include <cstdint>
#include <cstddef>

#define D_DIM 4096
#define M_DIM 8192
#define NLAYERS 4

typedef int v8i __attribute__((ext_vector_type(8)));
typedef int vi4 __attribute__((ext_vector_type(4)));

#if defined(__HIP_DEVICE_COMPILE__)
#if __has_builtin(__builtin_amdgcn_global_load_async_to_lds_b128) && \
    __has_builtin(__builtin_amdgcn_s_wait_asynccnt)
#define BITNET_ASYNC 1
#endif
#endif

__device__ __forceinline__ void copy16(const int8_t* g, int8_t* l) {
#ifdef BITNET_ASYNC
    typedef __attribute__((address_space(1))) vi4 gvec_t;
    typedef __attribute__((address_space(3))) vi4 lvec_t;
    __builtin_amdgcn_global_load_async_to_lds_b128(
        (gvec_t*)(uintptr_t)g, (lvec_t*)(uint32_t)(uintptr_t)l, 0, 0);
#else
    *(int4*)l = *(const int4*)g;
#endif
}

template <int N>
__device__ __forceinline__ void wait_copies() {
#ifdef BITNET_ASYNC
    __builtin_amdgcn_s_wait_asynccnt(N);
#endif
}

__device__ __forceinline__ int clamp_i8(int v) {
    return v < -128 ? -128 : (v > 127 ? 127 : v);
}

__global__ void __launch_bounds__(256) quant_weight_kernel(
    const float* __restrict__ Ws, int8_t* __restrict__ Bq,
    const float* __restrict__ w_scales)
{
    __shared__ float tile[32][33];
    const int l = blockIdx.z;
    const float* W = Ws + (size_t)l * D_DIM * D_DIM;
    int8_t* B = Bq + (size_t)l * D_DIM * D_DIM;
    const float inv = 1.0f / w_scales[l];
    const int k0 = blockIdx.x * 32, n0 = blockIdx.y * 32;
    const int tx = threadIdx.x, ty = threadIdx.y;
#pragma unroll
    for (int j = 0; j < 4; ++j) {
        const int nn = n0 + ty + j * 8;
        tile[ty + j * 8][tx] = W[(size_t)nn * D_DIM + k0 + tx];
    }
    __syncthreads();
#pragma unroll
    for (int j = 0; j < 4; ++j) {
        const int kk = k0 + ty + j * 8;
        int q = __float2int_rn(tile[tx][ty + j * 8] * inv);
        q = q > 1 ? 1 : (q < -1 ? -1 : q);
        B[(size_t)kk * D_DIM + n0 + tx] = (int8_t)q;
    }
}

__global__ void __launch_bounds__(256) quant_act_kernel(
    const float* __restrict__ X, int8_t* __restrict__ Xq,
    const float* __restrict__ in_scales, size_t n4)
{
    const float qs = 1.0f / in_scales[0];
    const size_t i = ((size_t)blockIdx.x * 256 + threadIdx.x) * 4;
    if (i + 3 < n4 * 4) {
        float4 v = *(const float4*)(X + i);
        char4 o;
        o.x = (char)clamp_i8(__float2int_rn(v.x * qs));
        o.y = (char)clamp_i8(__float2int_rn(v.y * qs));
        o.z = (char)clamp_i8(__float2int_rn(v.z * qs));
        o.w = (char)clamp_i8(__float2int_rn(v.w * qs));
        *(char4*)(Xq + i) = o;
    }
}

__global__ void __launch_bounds__(256, 2) bitnet_gemm_kernel(
    const int8_t* __restrict__ Aq, const int8_t* __restrict__ Bq,
    float* __restrict__ Out,
    const float* __restrict__ in_scales, const float* __restrict__ w_scales,
    int layer)
{
    __shared__ __align__(16) int8_t As[2][128 * 64];
    __shared__ __align__(16) int8_t Bs[2][64 * 128];

    const int t  = threadIdx.x;
    const int m0 = blockIdx.y * 128;
    const int n0 = blockIdx.x * 128;

    const int arow = t >> 2, aoff = (t & 3) << 4;
    const int brow = t >> 3, boff = (t & 7) << 4;

    const int lane = t & 31;
    const int wave = t >> 5;
    const int wm   = (wave & 3) << 5;
    const int wn   = (wave >> 2) << 6;
    const int half = lane >> 4;
    const int l16  = lane & 15;

    v8i acc[2][4] = {};

    auto load_tiles = [&](int buf, int kt) {
        const int8_t* ag = Aq + (size_t)(m0 + arow) * D_DIM + (kt << 6) + aoff;
        const int8_t* bg = Bq + ((size_t)(kt << 6) + brow) * D_DIM + n0 + boff;
        copy16(ag,                          &As[buf][arow * 64 + aoff]);
        copy16(ag + (size_t)64 * D_DIM,     &As[buf][(arow + 64) * 64 + aoff]);
        copy16(bg,                          &Bs[buf][brow * 128 + boff]);
        copy16(bg + (size_t)32 * D_DIM,     &Bs[buf][(brow + 32) * 128 + boff]);
    };

    load_tiles(0, 0);

    const int KSTEPS = D_DIM / 64;
    for (int kt = 0; kt < KSTEPS; ++kt) {
        const int buf = kt & 1;
        if (kt + 1 < KSTEPS) { load_tiles(buf ^ 1, kt + 1); wait_copies<4>(); }
        else                 { wait_copies<0>(); }
        __syncthreads();

        v8i a[2], b[4];
#pragma unroll
        for (int mi = 0; mi < 2; ++mi) {
            const int8_t* ap = &As[buf][(wm + mi * 16 + l16) * 64 + half * 8];
            int2 p0 = *(const int2*)(ap);
            int2 p1 = *(const int2*)(ap + 16);
            int2 p2 = *(const int2*)(ap + 32);
            int2 p3 = *(const int2*)(ap + 48);
            a[mi][0] = p0.x; a[mi][1] = p0.y; a[mi][2] = p1.x; a[mi][3] = p1.y;
            a[mi][4] = p2.x; a[mi][5] = p2.y; a[mi][6] = p3.x; a[mi][7] = p3.y;
        }
#pragma unroll
        for (int ni = 0; ni < 4; ++ni) {
            const uint8_t* bp = (const uint8_t*)&Bs[buf][wn + ni * 16 + l16];
#pragma unroll
            for (int r = 0; r < 8; ++r) {
                const int kb = (r >> 1) * 16 + half * 8 + (r & 1) * 4;
                b[ni][r] = (int)((uint32_t)bp[kb * 128] | ((uint32_t)bp[(kb + 1) * 128] << 8) |
                                 ((uint32_t)bp[(kb + 2) * 128] << 16) | ((uint32_t)bp[(kb + 3) * 128] << 24));
            }
        }
#pragma unroll
        for (int mi = 0; mi < 2; ++mi)
#pragma unroll
            for (int ni = 0; ni < 4; ++ni)
                acc[mi][ni] = __builtin_amdgcn_wmma_i32_16x16x64_iu8(
                    true, a[mi], true, b[ni], acc[mi][ni], false, false);
        __syncthreads();
    }

    const float s = in_scales[layer] * w_scales[layer];
#pragma unroll
    for (int mi = 0; mi < 2; ++mi) {
#pragma unroll
        for (int ni = 0; ni < 4; ++ni) {
            const int gm = m0 + wm + mi * 16 + half * 8;
            const int gn = n0 + wn + ni * 16 + l16;
            float* op = Out + (size_t)gm * D_DIM + gn;
#pragma unroll
            for (int v = 0; v < 8; ++v)
                op[(size_t)v * D_DIM] = (float)acc[mi][ni][v] * s;
        }
    }
}

__global__ void __launch_bounds__(256) ln_relu_quant_kernel(
    const float* __restrict__ H, const float* __restrict__ gamma,
    const float* __restrict__ beta, int8_t* __restrict__ Xq,
    const float* __restrict__ in_scales, int next_layer)
{
    __shared__ float row[D_DIM];
    __shared__ float s1[256];
    __shared__ float s2[256];
    const int r = blockIdx.x, t = threadIdx.x;
    const float* hp = H + (size_t)r * D_DIM;

    float sum = 0.0f, sq = 0.0f;
    for (int i = t; i < D_DIM; i += 256) {
        const float h = fmaxf(hp[i], 0.0f);
        row[i] = h;
        sum += h;
        sq  += h * h;
    }
    s1[t] = sum; s2[t] = sq;
    __syncthreads();
    for (int st = 128; st > 0; st >>= 1) {
        if (t < st) { s1[t] += s1[t + st]; s2[t] += s2[t + st]; }
        __syncthreads();
    }
    const float mu  = s1[0] * (1.0f / D_DIM);
    const float var = s2[0] * (1.0f / D_DIM) - mu * mu;
    const float rs  = rsqrtf(var + 1e-5f);
    const float qs  = 1.0f / in_scales[next_layer];

    for (int i = t; i < D_DIM; i += 256) {
        const float y = (row[i] - mu) * rs * gamma[i] + beta[i];
        Xq[(size_t)r * D_DIM + i] = (int8_t)clamp_i8(__float2int_rn(y * qs));
    }
}

extern "C" void kernel_launch(void* const* d_in, const int* in_sizes, int n_in,
                              void* d_out, int out_size, void* d_ws, size_t ws_size,
                              hipStream_t stream)
{
    const float* x         = (const float*)d_in[0];
    const float* Ws        = (const float*)d_in[1];
    const float* w_scales  = (const float*)d_in[2];
    const float* in_scales = (const float*)d_in[3];
    const float* gammas    = (const float*)d_in[4];
    const float* betas     = (const float*)d_in[5];
    float* out             = (float*)d_out;

    const size_t DSZ = (size_t)D_DIM * D_DIM;
    const size_t XSZ = (size_t)M_DIM * D_DIM;

    int8_t* Bq  = (int8_t*)d_ws;
    int8_t* Xq0 = Bq + (size_t)NLAYERS * DSZ;
    int8_t* Xq1 = Xq0 + XSZ;
    float*  H   = (float*)(Xq1 + XSZ);

    quant_weight_kernel<<<dim3(D_DIM / 32, D_DIM / 32, NLAYERS), dim3(32, 8), 0, stream>>>(
        Ws, Bq, w_scales);
    quant_act_kernel<<<(unsigned)(XSZ / 4 / 256), 256, 0, stream>>>(
        x, Xq0, in_scales, XSZ / 4);

    int8_t* xa = Xq0;
    int8_t* xb = Xq1;
    for (int l = 0; l < NLAYERS; ++l) {
        float* o = (l == NLAYERS - 1) ? out : H;
        bitnet_gemm_kernel<<<dim3(D_DIM / 128, M_DIM / 128), 256, 0, stream>>>(
            xa, Bq + (size_t)l * DSZ, o, in_scales, w_scales, l);
        if (l < NLAYERS - 1) {
            ln_relu_quant_kernel<<<M_DIM, 256, 0, stream>>>(
                H, gammas + (size_t)l * D_DIM, betas + (size_t)l * D_DIM,
                xb, in_scales, l + 1);
            int8_t* tmp = xa; xa = xb; xb = tmp;
        }
    }
    (void)in_sizes; (void)n_in; (void)out_size; (void)ws_size;
}
